// StableMultiheadAttention_22600117912081
// MI455X (gfx1250) — matrix operand order repaired and checked by reading, not compiled, not run on hardware
//
#include <hip/hip_runtime.h>
#include <hip/hip_bf16.h>

// ---------------------------------------------------------------------------
// StableMultiheadAttention for MI455X (gfx1250, wave32, WMMA)
//
// All four matmuls on v_wmma_f32_16x16x32_bf16 (fp32 accumulate); fp32
// flash-softmax (base-2 domain, log2e folded into the Q scale).
//
//   0) convert x, qkv_w, out_w  fp32 -> bf16        (vectorized)
//   1) QKV GEMM (32x64 per wave, double-buffered, prefetch) -> Q*, K, V^T
//   2) flash attention, 64-key blocks, 16 WMMA/iter; V frags loaded before
//      the softmax so their latency hides under the VALU work; LDS P-bounce
//   3) out projection (32x64 per wave, double-buffered) -> fp32 + bias
// ---------------------------------------------------------------------------

#define BATCH 4
#define SEQ   2048
#define DM    1024
#define NH    16
#define HD    64
#define ROWS  (BATCH * SEQ)   // 8192

typedef __attribute__((ext_vector_type(16))) __bf16 v16bf;
typedef __attribute__((ext_vector_type(8)))  float  v8f;
typedef __attribute__((ext_vector_type(4)))  unsigned int v4u;
typedef __attribute__((ext_vector_type(4)))  unsigned short v4us;

union FragU { v16bf v; v4u u[2]; };

__device__ __forceinline__ unsigned short f32_to_bf16(float f) {
  unsigned int u = __float_as_uint(f);
  u += 0x7FFFu + ((u >> 16) & 1u);   // round-to-nearest-even
  return (unsigned short)(u >> 16);
}

__device__ __forceinline__ v4u ld_v4u(const unsigned short* p) {
  return *reinterpret_cast<const v4u*>(p);
}

__device__ __forceinline__ v8f wmma_bf16(v16bf a, v16bf b, v8f c) {
  return __builtin_amdgcn_wmma_f32_16x16x32_bf16(
      false, a, false, b, (short)0, c, false, false);
}

// ---------------------------------------------------------------------------
// Kernel 0: fp32 -> bf16, 4 elements per thread (b128 in, b64 out)
// ---------------------------------------------------------------------------
__global__ __launch_bounds__(256) void cvt_f32_bf16(
    const float4* __restrict__ in, v4us* __restrict__ out, int n4) {
  int i = blockIdx.x * blockDim.x + threadIdx.x;
  int stride = gridDim.x * blockDim.x;
  for (; i < n4; i += stride) {
    float4 f = in[i];
    v4us o;
    o.x = f32_to_bf16(f.x); o.y = f32_to_bf16(f.y);
    o.z = f32_to_bf16(f.z); o.w = f32_to_bf16(f.w);
    out[i] = o;
  }
}

// ---------------------------------------------------------------------------
// Shared 32(M) x 64(N) double-buffered WMMA GEMM core with global_prefetch
// ~8 K-steps ahead.  Next-iter frags are loaded before the current WMMAs so
// the compiler emits partial s_wait_loadcnt (verified in round-2 disasm).
// ---------------------------------------------------------------------------
#define GEMM_CORE(ABASE, BBASE, KDIM)                                         \
  v8f acc[2][4] = {};                                                         \
  const unsigned short* arow0 = (ABASE) + (size_t)(mbase + lo) * (KDIM) + (hi ? 8 : 0); \
  const unsigned short* arow1 = arow0 + (size_t)16 * (KDIM);                  \
  const unsigned short* brow[4];                                              \
  _Pragma("unroll")                                                           \
  for (int t = 0; t < 4; ++t)                                                 \
    brow[t] = (BBASE) + (size_t)(nbase + t * 16 + lo) * (KDIM) + (hi ? 8 : 0); \
  FragU a0[2], a1[2], bb[2][4];                                               \
  a0[0].u[0] = ld_v4u(arow0);      a0[0].u[1] = ld_v4u(arow0 + 16);           \
  a1[0].u[0] = ld_v4u(arow1);      a1[0].u[1] = ld_v4u(arow1 + 16);           \
  _Pragma("unroll")                                                           \
  for (int t = 0; t < 4; ++t) {                                               \
    bb[0][t].u[0] = ld_v4u(brow[t]);                                          \
    bb[0][t].u[1] = ld_v4u(brow[t] + 16);                                     \
  }                                                                           \
  for (int k = 0; k < (KDIM); k += 32) {                                      \
    const int cur = (k >> 5) & 1, nxt = cur ^ 1;                              \
    const int kn = (k + 32 < (KDIM)) ? k + 32 : k;   /* clamped prefetch */   \
    a0[nxt].u[0] = ld_v4u(arow0 + kn);      a0[nxt].u[1] = ld_v4u(arow0 + kn + 16); \
    a1[nxt].u[0] = ld_v4u(arow1 + kn);      a1[nxt].u[1] = ld_v4u(arow1 + kn + 16); \
    _Pragma("unroll")                                                         \
    for (int t = 0; t < 4; ++t) {                                             \
      bb[nxt][t].u[0] = ld_v4u(brow[t] + kn);                                 \
      bb[nxt][t].u[1] = ld_v4u(brow[t] + kn + 16);                            \
    }                                                                         \
    if (k + 256 < (KDIM)) {   /* warm L2/WGP$ ~8 iterations ahead */          \
      __builtin_prefetch(arow0 + k + 256, 0, 3);                              \
      __builtin_prefetch(arow1 + k + 256, 0, 3);                              \
      _Pragma("unroll")                                                       \
      for (int t = 0; t < 4; ++t) __builtin_prefetch(brow[t] + k + 256, 0, 3);\
    }                                                                         \
    _Pragma("unroll")                                                         \
    for (int t = 0; t < 4; ++t)                                               \
      acc[0][t] = wmma_bf16(a0[cur].v, bb[cur][t].v, acc[0][t]);              \
    _Pragma("unroll")                                                         \
    for (int t = 0; t < 4; ++t)                                               \
      acc[1][t] = wmma_bf16(a1[cur].v, bb[cur][t].v, acc[1][t]);              \
  }

// ---------------------------------------------------------------------------
// Kernel 1: QKV projection.  C[8192,3072] = Xbf * Wbf^T + bias.
// Epilogue scatters into Q [b,h,s,hd] (scaled by hd^-0.5 * log2e), K
// [b,h,s,hd], V^T [b,h,hd,s] (bf16) so attention WMMA operands load
// contiguously per lane.
// ---------------------------------------------------------------------------
__global__ __launch_bounds__(256) void qkv_gemm(
    const unsigned short* __restrict__ xbf,   // [8192][1024]
    const unsigned short* __restrict__ wbf,   // [3072][1024]
    const float* __restrict__ bias,           // [3072]
    unsigned short* __restrict__ qbuf,        // [4][16][2048][64]
    unsigned short* __restrict__ kbuf,        // [4][16][2048][64]
    unsigned short* __restrict__ vtbuf)       // [4][16][64][2048]
{
  const int lane  = threadIdx.x & 31;
  const int wave  = threadIdx.x >> 5;
  const int wtile = blockIdx.x * 8 + wave;    // 256*48 = 12288 wave tiles
  const int mbase = (wtile & 255) * 32;
  const int nbase = (wtile >> 8) * 64;
  const int hi = lane >> 4, lo = lane & 15;
  const float QSCALE = 0.125f * 1.44269504088896f;  // hd^-0.5 * log2(e)

  GEMM_CORE(xbf, wbf, DM)

#pragma unroll
  for (int half = 0; half < 2; ++half) {
#pragma unroll
    for (int t = 0; t < 4; ++t) {
      const int col = nbase + t * 16 + lo;
      const float bv = bias[col];
#pragma unroll
      for (int r = 0; r < 8; ++r) {
        const int row = mbase + half * 16 + r + hi * 8;
        const float v = acc[half][t][r] + bv;
        const int b = row >> 11, s = row & (SEQ - 1);
        if (col < DM) {                                 // Q (scale folded in)
          const int h = col >> 6, d = col & 63;
          qbuf[((size_t)(b * NH + h) * SEQ + s) * HD + d] = f32_to_bf16(v * QSCALE);
        } else if (col < 2 * DM) {                      // K
          const int c = col - DM, h = c >> 6, d = c & 63;
          kbuf[((size_t)(b * NH + h) * SEQ + s) * HD + d] = f32_to_bf16(v);
        } else {                                        // V, transposed
          const int c = col - 2 * DM, h = c >> 6, d = c & 63;
          vtbuf[((size_t)(b * NH + h) * HD + d) * SEQ + s] = f32_to_bf16(v);
        }
      }
    }
  }
}

// ---------------------------------------------------------------------------
// Kernel 2: flash attention, one wave per (b, h, 16-row q tile), 64-key
// blocks: 8 WMMA scores -> issue all V-fragment loads -> base-2 online
// softmax (latency of V loads hides under the VALU work) -> LDS P-bounce
// (C-layout -> A-fragment layout; same-wave DS ops are in-order, ISA §7.3)
// -> 8 WMMA for O += P*V.  Next key-block warmed via global_prefetch.
// ---------------------------------------------------------------------------
__global__ __launch_bounds__(256) void attn(
    const unsigned short* __restrict__ qbuf,
    const unsigned short* __restrict__ kbuf,
    const unsigned short* __restrict__ vtbuf,
    unsigned short* __restrict__ aobf)         // [8192][1024] bf16
{
  __shared__ unsigned short plds[8][16 * 64];  // 2KB per wave
  const int lane = threadIdx.x & 31;
  const int wave = threadIdx.x >> 5;
  const int wid  = blockIdx.x * 8 + wave;      // 8192 waves
  const int qt = wid & 127;
  const int h  = (wid >> 7) & (NH - 1);
  const int b  = wid >> 11;
  const int hi = lane >> 4, lo = lane & 15;
  const int qbase = qt * 16;

  const unsigned short* Q  = qbuf  + (size_t)(b * NH + h) * SEQ * HD;
  const unsigned short* K  = kbuf  + (size_t)(b * NH + h) * SEQ * HD;
  const unsigned short* VT = vtbuf + (size_t)(b * NH + h) * HD * SEQ;

  // Q A-fragments for hd 0..31 and 32..63 (resident for the whole loop)
  FragU qf0, qf1;
  {
    const unsigned short* qrow = Q + (size_t)(qbase + lo) * HD;
    qf0.u[0] = ld_v4u(qrow + (hi ? 8 : 0));
    qf0.u[1] = ld_v4u(qrow + 16 + (hi ? 8 : 0));
    qf1.u[0] = ld_v4u(qrow + 32 + (hi ? 8 : 0));
    qf1.u[1] = ld_v4u(qrow + 48 + (hi ? 8 : 0));
  }

  v8f o[4] = {};
  float mrun[8], lrun[8];
#pragma unroll
  for (int r = 0; r < 8; ++r) { mrun[r] = -1e30f; lrun[r] = 0.0f; }

  unsigned short* P = &plds[wave][0];

  for (int kb = 0; kb < SEQ; kb += 64) {
    // ---- scores: four 16x16 tiles (keys kb+16j), base-2 domain ----
    v8f c[4] = {};
#pragma unroll
    for (int j = 0; j < 4; ++j) {
      const unsigned short* kr = K + (size_t)(kb + j * 16 + lo) * HD + (hi ? 8 : 0);
      FragU kf;
      kf.u[0] = ld_v4u(kr);      kf.u[1] = ld_v4u(kr + 16);   // hd 0..31
      c[j] = wmma_bf16(qf0.v, kf.v, c[j]);
      kf.u[0] = ld_v4u(kr + 32); kf.u[1] = ld_v4u(kr + 48);   // hd 32..63
      c[j] = wmma_bf16(qf1.v, kf.v, c[j]);
    }

    // ---- issue ALL V-fragment loads now: independent of the softmax, so
    //      their L2 latency hides under the ~200 VALU ops below (the later
    //      asm memory clobbers would otherwise pin them after the bounce) ----
    FragU vf[4][2];
#pragma unroll
    for (int t = 0; t < 4; ++t) {
#pragma unroll
      for (int kk = 0; kk < 2; ++kk) {
        const unsigned short* vrow =
            VT + (size_t)(t * 16 + lo) * SEQ + kb + kk * 32 + (hi ? 8 : 0);
        vf[t][kk].u[0] = ld_v4u(vrow);
        vf[t][kk].u[1] = ld_v4u(vrow + 16);
      }
    }

    // ---- warm next key block (K rows = 128B each; V^T spans) ----
    if (kb + 64 < SEQ) {
#pragma unroll
      for (int j = 0; j < 4; ++j)
        __builtin_prefetch(K + (size_t)(kb + 64 + j * 16 + lo) * HD, 0, 3);
#pragma unroll
      for (int t = 0; t < 4; ++t)
        __builtin_prefetch(VT + (size_t)(t * 16 + lo) * SEQ + kb + 64, 0, 3);
    }

    // ---- online softmax over 64 new columns (row = r + 8*hi) ----
#pragma unroll
    for (int r = 0; r < 8; ++r) {
      float tmax = fmaxf(fmaxf(c[0][r], c[1][r]), fmaxf(c[2][r], c[3][r]));
      tmax = fmaxf(tmax, __shfl_xor(tmax, 1, 32));
      tmax = fmaxf(tmax, __shfl_xor(tmax, 2, 32));
      tmax = fmaxf(tmax, __shfl_xor(tmax, 4, 32));
      tmax = fmaxf(tmax, __shfl_xor(tmax, 8, 32));
      const float mnew  = fmaxf(mrun[r], tmax);
      const float alpha = exp2f(mrun[r] - mnew);
      float p0 = exp2f(c[0][r] - mnew);
      float p1 = exp2f(c[1][r] - mnew);
      float p2 = exp2f(c[2][r] - mnew);
      float p3 = exp2f(c[3][r] - mnew);
      c[0][r] = p0; c[1][r] = p1; c[2][r] = p2; c[3][r] = p3;
      float ls = (p0 + p1) + (p2 + p3);
      ls += __shfl_xor(ls, 1, 32);
      ls += __shfl_xor(ls, 2, 32);
      ls += __shfl_xor(ls, 4, 32);
      ls += __shfl_xor(ls, 8, 32);
      lrun[r] = lrun[r] * alpha + ls;
      mrun[r] = mnew;
      o[0][r] *= alpha; o[1][r] *= alpha; o[2][r] *= alpha; o[3][r] *= alpha;
    }

    // ---- P: C layout -> LDS (row-major 16x64 bf16) -> A-fragments ----
#pragma unroll
    for (int r = 0; r < 8; ++r) {
      const int row = r + 8 * hi;
#pragma unroll
      for (int j = 0; j < 4; ++j)
        P[row * 64 + j * 16 + lo] = f32_to_bf16(c[j][r]);
    }
    __asm__ volatile("" ::: "memory");  // order DS load-after-store (same wave)
    FragU pf[2];
#pragma unroll
    for (int kk = 0; kk < 2; ++kk) {    // key sub-blocks kb+32*kk
      const unsigned short* prow = P + lo * 64 + kk * 32;
      pf[kk].u[0] = ld_v4u(prow + (hi ? 8 : 0));
      pf[kk].u[1] = ld_v4u(prow + 16 + (hi ? 8 : 0));
    }
    __asm__ volatile("" ::: "memory");  // keep next iter's stores behind these loads

    // ---- O += P(16x64) * V(64x64) with preloaded V fragments ----
#pragma unroll
    for (int t = 0; t < 4; ++t) {
#pragma unroll
      for (int kk = 0; kk < 2; ++kk)
        o[t] = wmma_bf16(pf[kk].v, vf[t][kk].v, o[t]);
    }
  }

  // ---- normalize and store (merge heads) ----
#pragma unroll
  for (int r = 0; r < 8; ++r) {
    const int s = qbase + r + 8 * hi;
    const float inv = 1.0f / lrun[r];
#pragma unroll
    for (int t = 0; t < 4; ++t) {
      const int d = h * HD + t * 16 + lo;
      aobf[((size_t)b * SEQ + s) * DM + d] = f32_to_bf16(o[t][r] * inv);
    }
  }
}

// ---------------------------------------------------------------------------
// Kernel 3: output projection.  y[8192,1024] = AO * Wout^T + bias (fp32 out)
// ---------------------------------------------------------------------------
__global__ __launch_bounds__(256) void out_gemm(
    const unsigned short* __restrict__ aobf,  // [8192][1024]
    const unsigned short* __restrict__ wbf,   // [1024][1024]
    const float* __restrict__ bias,           // [1024]
    float* __restrict__ y)                    // [8192][1024]
{
  const int lane  = threadIdx.x & 31;
  const int wave  = threadIdx.x >> 5;
  const int wtile = blockIdx.x * 8 + wave;    // 256*16 = 4096 wave tiles
  const int mbase = (wtile & 255) * 32;
  const int nbase = (wtile >> 8) * 64;
  const int hi = lane >> 4, lo = lane & 15;

  GEMM_CORE(aobf, wbf, DM)

#pragma unroll
  for (int half = 0; half < 2; ++half) {
#pragma unroll
    for (int t = 0; t < 4; ++t) {
      const int col = nbase + t * 16 + lo;
      const float bv = bias[col];
#pragma unroll
      for (int r = 0; r < 8; ++r) {
        const int row = mbase + half * 16 + r + 8 * hi;
        y[(size_t)row * DM + col] = acc[half][t][r] + bv;
      }
    }
  }
}

// ---------------------------------------------------------------------------
extern "C" void kernel_launch(void* const* d_in, const int* in_sizes, int n_in,
                              void* d_out, int out_size, void* d_ws, size_t ws_size,
                              hipStream_t stream) {
  const float* x      = (const float*)d_in[0];
  const float* qkv_w  = (const float*)d_in[1];
  const float* qkv_b  = (const float*)d_in[2];
  const float* out_w  = (const float*)d_in[3];
  const float* out_b  = (const float*)d_in[4];
  float* out = (float*)d_out;

  unsigned char* ws = (unsigned char*)d_ws;
  size_t off = 0;
  auto take = [&](size_t bytes) {
    void* p = ws + off;
    off += (bytes + 255) & ~(size_t)255;
    return p;
  };
  unsigned short* xbf  = (unsigned short*)take((size_t)ROWS * DM * 2);
  unsigned short* wqkv = (unsigned short*)take((size_t)3 * DM * DM * 2);
  unsigned short* wout = (unsigned short*)take((size_t)DM * DM * 2);
  unsigned short* qbuf = (unsigned short*)take((size_t)ROWS * DM * 2);
  unsigned short* kbuf = (unsigned short*)take((size_t)ROWS * DM * 2);
  unsigned short* vtbf = (unsigned short*)take((size_t)ROWS * DM * 2);
  unsigned short* aobf = (unsigned short*)take((size_t)ROWS * DM * 2);

  cvt_f32_bf16<<<2048, 256, 0, stream>>>((const float4*)x,     (v4us*)xbf,  ROWS * DM / 4);
  cvt_f32_bf16<<<1024, 256, 0, stream>>>((const float4*)qkv_w, (v4us*)wqkv, 3 * DM * DM / 4);
  cvt_f32_bf16<<< 512, 256, 0, stream>>>((const float4*)out_w, (v4us*)wout, DM * DM / 4);

  // 256 M-tiles x 48 N-tiles = 12288 wave tiles / 8 waves per block
  qkv_gemm<<<1536, 256, 0, stream>>>(xbf, wqkv, qkv_b, qbuf, kbuf, vtbf);
  // 4 batch x 16 heads x 128 q-tiles = 8192 waves / 8 per block
  attn<<<1024, 256, 0, stream>>>(qbuf, kbuf, vtbf, aobf);
  // 256 M-tiles x 16 N-tiles = 4096 wave tiles / 8 per block
  out_gemm<<<512, 256, 0, stream>>>(aobf, wout, out_b, out);
}
